// Cobrablock_46076409151775
// MI455X (gfx1250) — matrix operand order repaired and checked by reading, not compiled, not run on hardware
//
#include <hip/hip_runtime.h>
#include <hip/hip_bf16.h>
#include <stdint.h>

#define D_MODEL 512
#define SEQ_LEN 8192
#define BATCH   4
#define NROWS   (BATCH * SEQ_LEN)      // 32768
#define EPS_LN  1e-5f
#define CHUNK   256
#define NCHUNK  (SEQ_LEN / CHUNK)      // 32

typedef __bf16 bf16_t;
typedef bf16_t v16bf __attribute__((ext_vector_type(16)));
typedef float  v8f   __attribute__((ext_vector_type(8)));
typedef int    v4i_async __attribute__((vector_size(16)));   // pointee type of async-LDS builtin

union BFrag { v16bf v; uint4 q[2]; };
union AccU  { v8f v; float f[8]; };

__device__ __forceinline__ uint16_t f2bf(float f) {
    uint32_t u = __float_as_uint(f);
    uint32_t r = u + 0x7FFFu + ((u >> 16) & 1u);   // round-to-nearest-even
    return (uint16_t)(r >> 16);
}

// --- gfx1250 async global->LDS copy (16 bytes per lane), with asm fallback ---
__device__ __forceinline__ void async_copy16(const uint16_t* gp, uint16_t* lp) {
#if __has_builtin(__builtin_amdgcn_global_load_async_to_lds_b128)
    __builtin_amdgcn_global_load_async_to_lds_b128(
        (v4i_async*)(void*)(const_cast<uint16_t*>(gp)),
        (v4i_async*)(void*)lp, 0, 0);
#else
    asm volatile("global_load_async_to_lds_b128 %0, %1, off"
                 :: "v"(lp), "v"(gp)
                 : "memory");
#endif
}

__device__ __forceinline__ void async_wait0() {
#if __has_builtin(__builtin_amdgcn_s_wait_asynccnt)
    __builtin_amdgcn_s_wait_asynccnt(0);
#else
    asm volatile("s_wait_asynccnt 0" ::: "memory");
#endif
}

// ---------------- LayerNorm (one row of 512 per 128-thread block) ------------
template <bool BF16OUT>
__global__ void ln_kernel(const float* __restrict__ x,
                          const float* __restrict__ g,
                          const float* __restrict__ b,
                          void* __restrict__ dst) {
    __shared__ float red[128];
    const int row = blockIdx.x;
    const float* xr = x + (size_t)row * D_MODEL;
    float v[4];
    float s = 0.f;
#pragma unroll
    for (int i = 0; i < 4; ++i) { v[i] = xr[threadIdx.x + i * 128]; s += v[i]; }
    red[threadIdx.x] = s;
    __syncthreads();
    for (int st = 64; st > 0; st >>= 1) {
        if (threadIdx.x < st) red[threadIdx.x] += red[threadIdx.x + st];
        __syncthreads();
    }
    const float mean = red[0] * (1.f / D_MODEL);
    __syncthreads();
    float vs = 0.f;
#pragma unroll
    for (int i = 0; i < 4; ++i) { float d = v[i] - mean; vs += d * d; }
    red[threadIdx.x] = vs;
    __syncthreads();
    for (int st = 64; st > 0; st >>= 1) {
        if (threadIdx.x < st) red[threadIdx.x] += red[threadIdx.x + st];
        __syncthreads();
    }
    const float rstd = rsqrtf(red[0] * (1.f / D_MODEL) + EPS_LN);
#pragma unroll
    for (int i = 0; i < 4; ++i) {
        const int c = threadIdx.x + i * 128;
        const float o = g[c] * (v[i] - mean) * rstd + b[c];
        if (BF16OUT) ((uint16_t*)dst)[(size_t)row * D_MODEL + c] = f2bf(o);
        else         ((float*)dst)[(size_t)row * D_MODEL + c] = o;
    }
}

// ------------- Weight transpose + bf16 convert: dst[n*K+k] = src[k*N+n] -----
__global__ void wconv_kernel(const float* __restrict__ src, uint16_t* __restrict__ dst,
                             int K, int N) {
    const int idx = blockIdx.x * blockDim.x + threadIdx.x;
    if (idx >= K * N) return;
    const int n = idx / K, k = idx - n * K;
    dst[idx] = f2bf(src[(size_t)k * N + n]);
}

// ---------------- S4 chunked complex scan -----------------------------------
__global__ void s4_partial(const float* __restrict__ h,
                           const float* __restrict__ Ar, const float* __restrict__ Ai,
                           const float* __restrict__ Br, const float* __restrict__ Bi,
                           float* __restrict__ Sr, float* __restrict__ Si) {
    const int idx = blockIdx.x * blockDim.x + threadIdx.x;   // 0..65535
    const int c  = idx & (D_MODEL - 1);
    const int j  = (idx >> 9) & (NCHUNK - 1);
    const int bb = idx >> 14;
    const float ar = Ar[c], ai = Ai[c], br = Br[c], bi = Bi[c];
    float sr = 0.f, si = 0.f;
    const float* hp = h + ((size_t)(bb * SEQ_LEN + j * CHUNK)) * D_MODEL + c;
    for (int t = 0; t < CHUNK; ++t) {
        const float u = hp[(size_t)t * D_MODEL];
        const float nr = ar * sr - ai * si + br * u;
        const float ni = ar * si + ai * sr + bi * u;
        sr = nr; si = ni;
    }
    Sr[idx] = sr; Si[idx] = si;
}

__global__ void s4_combine(const float* __restrict__ Ar, const float* __restrict__ Ai,
                           const float* __restrict__ Sr, const float* __restrict__ Si,
                           float* __restrict__ Ir, float* __restrict__ Ii) {
    const int idx = blockIdx.x * blockDim.x + threadIdx.x;   // 0..2047
    const int c  = idx & (D_MODEL - 1);
    const int bb = idx >> 9;
    float pr = Ar[c], pi = Ai[c];
#pragma unroll
    for (int k = 0; k < 8; ++k) {                 // A^256 via 8 squarings
        const float nr = pr * pr - pi * pi;
        const float ni = 2.f * pr * pi;
        pr = nr; pi = ni;
    }
    float cr = 0.f, ci = 0.f;
    for (int j = 0; j < NCHUNK; ++j) {
        const int base = (bb * NCHUNK + j) * D_MODEL + c;
        Ir[base] = cr; Ii[base] = ci;
        const float nr = pr * cr - pi * ci + Sr[base];
        const float ni = pr * ci + pi * cr + Si[base];
        cr = nr; ci = ni;
    }
}

__global__ void s4_final(const float* __restrict__ h, const float* __restrict__ x,
                         const float* __restrict__ Ar, const float* __restrict__ Ai,
                         const float* __restrict__ Br, const float* __restrict__ Bi,
                         const float* __restrict__ Cr, const float* __restrict__ Ci,
                         const float* __restrict__ Dd,
                         const float* __restrict__ Ir, const float* __restrict__ Ii,
                         float* __restrict__ x1) {
    const int idx = blockIdx.x * blockDim.x + threadIdx.x;
    const int c  = idx & (D_MODEL - 1);
    const int j  = (idx >> 9) & (NCHUNK - 1);
    const int bb = idx >> 14;
    const float ar = Ar[c], ai = Ai[c], br = Br[c], bi = Bi[c];
    const float cr = Cr[c], ci = Ci[c], dd = Dd[c];
    float sr = Ir[idx], si = Ii[idx];
    const size_t base = ((size_t)(bb * SEQ_LEN + j * CHUNK)) * D_MODEL + c;
    for (int t = 0; t < CHUNK; ++t) {
        const size_t o = base + (size_t)t * D_MODEL;
        const float u = h[o];
        const float nr = ar * sr - ai * si + br * u;
        const float ni = ar * si + ai * sr + bi * u;
        sr = nr; si = ni;
        const float y = cr * sr - ci * si + dd * u;
        x1[o] = x[o] + y;
    }
}

// ---------------- GEMM1 (h2 @ W1 + b1) fused with SwiGLU --------------------
// grid: (NROWS/64, 512/32), block: 128 (4 waves).
// A-tiles (shared by all 4 waves) staged into LDS via async copies, double
// buffered; B-fragments loaded directly from transposed bf16 weights.
__global__ void gemm1_swiglu(const uint16_t* __restrict__ h2,
                             const uint16_t* __restrict__ W1t,   // [1024][512] bf16
                             const float* __restrict__ b1,
                             uint16_t* __restrict__ g_out) {
    __shared__ float ldsP[2][64][33];
    __shared__ __align__(16) uint16_t aT[2][64 * 32];
    const int m0   = blockIdx.x * 64;
    const int n0   = blockIdx.y * 32;
    const int wave = threadIdx.x >> 5;
    const int lane = threadIdx.x & 31;
    const int half = lane >> 4;
    const int ln   = lane & 15;
    const int part = wave >> 1;          // 0: gate input half, 1: linear half
    const int cg   = wave & 1;           // which 16 columns of the 32
    const int col  = (part ? D_MODEL : 0) + n0 + cg * 16 + ln;
    const uint16_t* wp = W1t + (size_t)col * D_MODEL;

    auto load_tile = [&](int buf, int k0) {
#pragma unroll
        for (int i = 0; i < 2; ++i) {
            const int chunk = threadIdx.x + i * 128;   // 0..255, 16B each
            const int row = chunk >> 2;
            const int c8  = (chunk & 3) * 8;
            async_copy16(h2 + (size_t)(m0 + row) * D_MODEL + k0 + c8,
                         &aT[buf][row * 32 + c8]);
        }
    };

    AccU acc[4];
#pragma unroll
    for (int t = 0; t < 4; ++t) acc[t].v = v8f{0, 0, 0, 0, 0, 0, 0, 0};

    load_tile(0, 0);
    for (int kt = 0; kt < D_MODEL / 32; ++kt) {
        const int k0 = kt * 32;
        async_wait0();
        __syncthreads();
        if (kt + 1 < D_MODEL / 32) load_tile((kt + 1) & 1, k0 + 32);
        const uint16_t* at = &aT[kt & 1][0];
        BFrag bfr;
        bfr.q[0] = *(const uint4*)(wp + k0 + half * 8);
        bfr.q[1] = *(const uint4*)(wp + k0 + half * 8 + 16);
        __builtin_prefetch(wp + k0 + 64, 0, 0);
#pragma unroll
        for (int t = 0; t < 4; ++t) {
            BFrag afr;
            const uint16_t* ap = at + (t * 16 + ln) * 32 + half * 8;
            afr.q[0] = *(const uint4*)(ap);
            afr.q[1] = *(const uint4*)(ap + 16);
            acc[t].v = __builtin_amdgcn_wmma_f32_16x16x32_bf16(
                false, afr.v, false, bfr.v, (short)0, acc[t].v, false, false);
        }
    }
    // Stash partials so the two halves can be paired across waves.
    __syncthreads();
#pragma unroll
    for (int t = 0; t < 4; ++t)
#pragma unroll
        for (int r = 0; r < 8; ++r)
            ldsP[part][t * 16 + r + half * 8][cg * 16 + ln] = acc[t].f[r];
    __syncthreads();

#pragma unroll
    for (int e = 0; e < 16; ++e) {
        const int idx = threadIdx.x + e * 128;   // 0..2047 over 64x32
        const int row = idx >> 5;
        const int cc  = idx & 31;
        const float v1 = ldsP[0][row][cc] + b1[n0 + cc];
        const float v2 = ldsP[1][row][cc] + b1[D_MODEL + n0 + cc];
        const float gv = (v1 / (1.f + __expf(-v1))) * v2;
        g_out[(size_t)(m0 + row) * D_MODEL + n0 + cc] = f2bf(gv);
    }
}

// ---------------- GEMM2 (g @ W2 + b2 + x1) ----------------------------------
// grid: (NROWS/64, 512/64), block: 128 (4 waves, 16 cols each).
__global__ void gemm2_out(const uint16_t* __restrict__ g,
                          const uint16_t* __restrict__ W2t,   // [512][512] bf16
                          const float* __restrict__ b2,
                          const float* __restrict__ x1,
                          float* __restrict__ out) {
    __shared__ __align__(16) uint16_t aT[2][64 * 32];
    const int m0   = blockIdx.x * 64;
    const int n0   = blockIdx.y * 64;
    const int wave = threadIdx.x >> 5;
    const int lane = threadIdx.x & 31;
    const int half = lane >> 4;
    const int ln   = lane & 15;
    const int col  = n0 + wave * 16 + ln;
    const uint16_t* wp = W2t + (size_t)col * D_MODEL;

    auto load_tile = [&](int buf, int k0) {
#pragma unroll
        for (int i = 0; i < 2; ++i) {
            const int chunk = threadIdx.x + i * 128;
            const int row = chunk >> 2;
            const int c8  = (chunk & 3) * 8;
            async_copy16(g + (size_t)(m0 + row) * D_MODEL + k0 + c8,
                         &aT[buf][row * 32 + c8]);
        }
    };

    AccU acc[4];
#pragma unroll
    for (int t = 0; t < 4; ++t) acc[t].v = v8f{0, 0, 0, 0, 0, 0, 0, 0};

    load_tile(0, 0);
    for (int kt = 0; kt < D_MODEL / 32; ++kt) {
        const int k0 = kt * 32;
        async_wait0();
        __syncthreads();
        if (kt + 1 < D_MODEL / 32) load_tile((kt + 1) & 1, k0 + 32);
        const uint16_t* at = &aT[kt & 1][0];
        BFrag bfr;
        bfr.q[0] = *(const uint4*)(wp + k0 + half * 8);
        bfr.q[1] = *(const uint4*)(wp + k0 + half * 8 + 16);
        __builtin_prefetch(wp + k0 + 64, 0, 0);
#pragma unroll
        for (int t = 0; t < 4; ++t) {
            BFrag afr;
            const uint16_t* ap = at + (t * 16 + ln) * 32 + half * 8;
            afr.q[0] = *(const uint4*)(ap);
            afr.q[1] = *(const uint4*)(ap + 16);
            acc[t].v = __builtin_amdgcn_wmma_f32_16x16x32_bf16(
                false, afr.v, false, bfr.v, (short)0, acc[t].v, false, false);
        }
    }
    const float bias = b2[col];
#pragma unroll
    for (int t = 0; t < 4; ++t)
#pragma unroll
        for (int r = 0; r < 8; ++r) {
            const int row = m0 + t * 16 + r + half * 8;
            const size_t o = (size_t)row * D_MODEL + col;
            out[o] = acc[t].f[r] + bias + x1[o];
        }
}

// -----------------------------------------------------------------------------
extern "C" void kernel_launch(void* const* d_in, const int* in_sizes, int n_in,
                              void* d_out, int out_size, void* d_ws, size_t ws_size,
                              hipStream_t stream) {
    const float* x     = (const float*)d_in[0];
    const float* ln1_g = (const float*)d_in[1];
    const float* ln1_b = (const float*)d_in[2];
    const float* Ar    = (const float*)d_in[3];
    const float* Ai    = (const float*)d_in[4];
    const float* Br    = (const float*)d_in[5];
    const float* Bi    = (const float*)d_in[6];
    const float* Cr    = (const float*)d_in[7];
    const float* Ci    = (const float*)d_in[8];
    const float* Dd    = (const float*)d_in[9];
    const float* ln2_g = (const float*)d_in[10];
    const float* ln2_b = (const float*)d_in[11];
    const float* W1    = (const float*)d_in[12];
    const float* b1    = (const float*)d_in[13];
    const float* W2    = (const float*)d_in[14];
    const float* b2    = (const float*)d_in[15];
    float* out = (float*)d_out;

    const size_t SZ = (size_t)NROWS * D_MODEL;  // 16M elements
    char* w = (char*)d_ws;
    float*    h    = (float*)(w);                     // 64 MB fp32
    float*    x1   = (float*)(w + SZ * 4);            // 64 MB fp32
    uint16_t* h2   = (uint16_t*)(w + SZ * 8);         // 32 MB bf16
    uint16_t* gbuf = (uint16_t*)(w + SZ * 10);        // 32 MB bf16
    uint16_t* W1t  = (uint16_t*)(w + SZ * 12);        // 1 MB
    uint16_t* W2t  = W1t + (size_t)D_MODEL * 2 * D_MODEL;
    float*    Sr   = (float*)(W2t + (size_t)D_MODEL * D_MODEL);
    float*    Si   = Sr + (size_t)BATCH * NCHUNK * D_MODEL;
    float*    Ir   = Si + (size_t)BATCH * NCHUNK * D_MODEL;
    float*    Ii   = Ir + (size_t)BATCH * NCHUNK * D_MODEL;

    // 1. LN1 -> h (fp32)
    ln_kernel<false><<<NROWS, 128, 0, stream>>>(x, ln1_g, ln1_b, (void*)h);
    // 2. Weight transpose + bf16 convert (independent of 1)
    wconv_kernel<<<(D_MODEL * 2 * D_MODEL + 255) / 256, 256, 0, stream>>>(W1, W1t, D_MODEL, 2 * D_MODEL);
    wconv_kernel<<<(D_MODEL * D_MODEL + 255) / 256, 256, 0, stream>>>(W2, W2t, D_MODEL, D_MODEL);
    // 3. S4 chunked scan + residual -> x1
    s4_partial<<<(BATCH * NCHUNK * D_MODEL) / 256, 256, 0, stream>>>(h, Ar, Ai, Br, Bi, Sr, Si);
    s4_combine<<<(BATCH * D_MODEL) / 256, 256, 0, stream>>>(Ar, Ai, Sr, Si, Ir, Ii);
    s4_final<<<(BATCH * NCHUNK * D_MODEL) / 256, 256, 0, stream>>>(h, x, Ar, Ai, Br, Bi, Cr, Ci, Dd, Ir, Ii, x1);
    // 4. LN2 -> h2 (bf16)
    ln_kernel<true><<<NROWS, 128, 0, stream>>>(x1, ln2_g, ln2_b, (void*)h2);
    // 5. GEMM1 + SwiGLU -> gbuf (bf16)
    gemm1_swiglu<<<dim3(NROWS / 64, D_MODEL / 32), 128, 0, stream>>>(h2, W1t, b1, gbuf);
    // 6. GEMM2 + bias + residual -> out (fp32)
    gemm2_out<<<dim3(NROWS / 64, D_MODEL / 64), 128, 0, stream>>>(gbuf, W2t, b2, x1, out);
}
